// GraphAttentionLayer_51565377356304
// MI455X (gfx1250) — matrix operand order repaired and checked by reading, not compiled, not run on hardware
//
#include <hip/hip_runtime.h>

#define LRELU_ALPHA 0.2f
#define IN_EPS 1e-5f

typedef __attribute__((ext_vector_type(16))) __bf16 v16bf;
typedef __attribute__((ext_vector_type(8)))  float  v8f;

// ---------------------------------------------------------------------------
// f32 -> bf16 with round-to-nearest-even
// ---------------------------------------------------------------------------
__device__ __forceinline__ __bf16 f2bf(float f) {
  union { float f; unsigned u; } v; v.f = f;
  unsigned r = v.u + 0x7FFFu + ((v.u >> 16) & 1u);
  unsigned short h = (unsigned short)(r >> 16);
  __bf16 b; __builtin_memcpy(&b, &h, 2); return b;
}

// ---------------------------------------------------------------------------
// 16x32 bf16 A-fragment from row-major f32.  Lane row = lane&15; halves
// 0..7 -> K = kb+0..7, halves 8..15 -> K = kb+16..23, kb = 8*(lane>>4).
// 4x global_load_b128 + pack.
// ---------------------------------------------------------------------------
__device__ __forceinline__ v16bf load_a_f32(const float* __restrict__ A,
                                            int lda, int row0, int k0) {
  int lane = threadIdx.x & 31;
  const float* p = A + (size_t)(row0 + (lane & 15)) * lda + k0 + ((lane >> 4) << 3);
  float4 x0 = *(const float4*)(p);
  float4 x1 = *(const float4*)(p + 4);
  float4 x2 = *(const float4*)(p + 16);
  float4 x3 = *(const float4*)(p + 20);
  v16bf a;
  a[0]=f2bf(x0.x); a[1]=f2bf(x0.y); a[2]=f2bf(x0.z); a[3]=f2bf(x0.w);
  a[4]=f2bf(x1.x); a[5]=f2bf(x1.y); a[6]=f2bf(x1.z); a[7]=f2bf(x1.w);
  a[8]=f2bf(x2.x); a[9]=f2bf(x2.y); a[10]=f2bf(x2.z); a[11]=f2bf(x2.w);
  a[12]=f2bf(x3.x); a[13]=f2bf(x3.y); a[14]=f2bf(x3.z); a[15]=f2bf(x3.w);
  return a;
}

// B fragments pre-packed lane-major: frag = 32 lanes x 16 halves (1 KB).
__device__ __forceinline__ v16bf load_b_pack(const __bf16* __restrict__ P,
                                             int kt, int ct, int nct) {
  int lane = threadIdx.x & 31;
  return *(const v16bf*)(P + (((size_t)(kt * nct + ct)) << 9) + (lane << 4));
}

// A-fragment straight out of bf16 LDS (row-major, padded stride).
__device__ __forceinline__ v16bf load_a_lds(const __bf16* s, int stride,
                                            int arow, int kb8, int kt) {
  const __bf16* pr = s + arow * stride + kt * 32 + kb8;
  v16bf a;
  #pragma unroll
  for (int j = 0; j < 8; ++j) { a[j] = pr[j]; a[8 + j] = pr[16 + j]; }
  return a;
}

// ---------------------------------------------------------------------------
// Repack row-major f32 weight [K,N] into WMMA B-fragment order (bf16).
// ---------------------------------------------------------------------------
__global__ void repack_b_kernel(const float* __restrict__ src,
                                __bf16* __restrict__ dst, int K, int N) {
  int e = blockIdx.x * 256 + threadIdx.x;
  if (e >= K * N) return;
  int nct = N >> 4;
  int f = e >> 9, h = e & 511;
  int l = h >> 4, j = h & 15;
  int kt = f / nct, ct = f - kt * nct;
  int k   = kt * 32 + ((j >> 3) << 4) + ((l >> 4) << 3) + (j & 7);
  int col = ct * 16 + (l & 15);
  dst[e] = f2bf(src[k * N + col]);
}

// Wa2[k] = sum_c W[k][c] * a[256+c]   (s3 = h_3d_row . Wa2; wh_3d never built)
__global__ void wa2_kernel(const float* __restrict__ W,
                           const float* __restrict__ a,
                           float* __restrict__ Wa2) {
  int k = threadIdx.x;
  float s = 0.f;
  for (int c = 0; c < 256; ++c) s += W[k * 256 + c] * a[256 + c];
  Wa2[k] = s;
}

// ---------------------------------------------------------------------------
// Fused attention: wh = h_2d_block @ W (WMMA, 128 rows = 16 nodes) -> LDS,
// scores, leaky-relu, softmax over 8 leaves, weighted agg, ELU -> h_prime.
// ---------------------------------------------------------------------------
__global__ __launch_bounds__(256) void gat_attn_kernel(
    const float* __restrict__ h2d, const float* __restrict__ h3d,
    const __bf16* __restrict__ Wpack, const float* __restrict__ a_vec,
    const float* __restrict__ Wa2, float* __restrict__ h_prime) {
  const int WHS = 260;
  __shared__ float s_wh[128 * 260];
  __shared__ float s_s2[128];
  __shared__ float s_s3[16];

  int blk = blockIdx.x;
  int row0  = blk * 128;
  int node0 = blk * 16;
  int tid = threadIdx.x;
  int w = tid >> 5;
  int lane = tid & 31;
  int mrow = row0 + w * 16;

  // --- Stage 1: software-pipelined WMMA, K=256 (8 k-tiles), N=256 -------
  v8f c[16] = {};
  v16bf a_cur = load_a_f32(h2d, 256, mrow, 0);
  v16bf a_nxt = a_cur;
  v16bf b_cur[4], b_nxt[4];
  #pragma unroll
  for (int t = 0; t < 4; ++t) b_cur[t] = load_b_pack(Wpack, 0, t, 16);

  #pragma unroll
  for (int gi = 0; gi < 32; ++gi) {          // 8 kt x 4 groups of 4 tiles
    const int kt = gi >> 2, g = gi & 3;
    if (g == 0 && kt + 1 < 8)
      a_nxt = load_a_f32(h2d, 256, mrow, (kt + 1) * 32);
    if (gi + 1 < 32) {
      const int nkt = (g == 3) ? kt + 1 : kt;
      const int ng  = (g == 3) ? 0 : g + 1;
      #pragma unroll
      for (int t = 0; t < 4; ++t)
        b_nxt[t] = load_b_pack(Wpack, nkt, ng * 4 + t, 16);
    }
    #pragma unroll
    for (int t = 0; t < 4; ++t)
      c[g * 4 + t] = __builtin_amdgcn_wmma_f32_16x16x32_bf16(
          false, a_cur, false, b_cur[t], (short)0, c[g * 4 + t], false, false);
    #pragma unroll
    for (int t = 0; t < 4; ++t) b_cur[t] = b_nxt[t];
    if (g == 3) a_cur = a_nxt;
  }

  // --- Stage 2: C frags -> LDS (M = i + 8*(lane>>4), N = lane&15) -------
  int rl = w * 16 + 8 * (lane >> 4);
  int cl = lane & 15;
  #pragma unroll
  for (int t = 0; t < 16; ++t)
    #pragma unroll
    for (int i = 0; i < 8; ++i)
      s_wh[(rl + i) * WHS + t * 16 + cl] = c[t][i];
  __syncthreads();

  // --- Stage 3: per-leaf score s2 and per-node score s3 -----------------
  if (tid < 128) {
    float s = 0.f;
    #pragma unroll 4
    for (int cc = 0; cc < 256; ++cc) s += s_wh[tid * WHS + cc] * a_vec[cc];
    s_s2[tid] = s;
  } else if (tid < 144) {
    int j = tid - 128;
    const float* hp = h3d + (size_t)(node0 + j) * 256;
    float s = 0.f;
    #pragma unroll 4
    for (int cc = 0; cc < 256; ++cc) s += hp[cc] * Wa2[cc];
    s_s3[j] = s;
  }
  __syncthreads();

  // --- Stage 4: softmax over 8 leaves + weighted aggregation + ELU ------
  int j  = tid >> 4;
  int c0 = tid & 15;
  float s3 = s_s3[j];
  float e[8], m = -3.4e38f;
  #pragma unroll
  for (int l = 0; l < 8; ++l) {
    float x = s3 + s_s2[j * 8 + l];
    x = x > 0.f ? x : LRELU_ALPHA * x;
    e[l] = x; m = fmaxf(m, x);
  }
  float sum = 0.f;
  #pragma unroll
  for (int l = 0; l < 8; ++l) { e[l] = __expf(e[l] - m); sum += e[l]; }
  float inv = 1.f / sum;
  float* outp = h_prime + (size_t)(node0 + j) * 256;
  for (int cc = c0; cc < 256; cc += 16) {
    float acc = 0.f;
    #pragma unroll
    for (int l = 0; l < 8; ++l) acc += e[l] * s_wh[(j * 8 + l) * WHS + cc];
    acc *= inv;
    acc = acc > 0.f ? acc : __expf(acc) - 1.f;
    outp[cc] = acc;
  }
}

// ---------------------------------------------------------------------------
// Fused MLP: y1 = relu([h3d|h_prime] @ W1) as bf16 in LDS (A-frag ready),
// y2 = y1 @ W2, InstanceNorm(256) + residual, single output write.
// ---------------------------------------------------------------------------
__device__ __forceinline__ v16bf load_a_mlp(const float* __restrict__ h3d,
                                            const float* __restrict__ hp,
                                            int mrow, int kt) {
  return (kt < 8) ? load_a_f32(h3d, 256, mrow, kt * 32)
                  : load_a_f32(hp,  256, mrow, (kt - 8) * 32);
}

__global__ __launch_bounds__(256) void gat_mlp_kernel(
    const float* __restrict__ h3d, const float* __restrict__ h_prime,
    const __bf16* __restrict__ W1pack, const __bf16* __restrict__ W2pack,
    float* __restrict__ out) {
  const int Y1S = 520;                        // bf16 stride (1040 B / row)
  const int Y2S = 260;                        // f32 stride (same bytes / row)
  __shared__ float smem[128 * 260];           // aliased y1(bf16) / y2(f32)
  __shared__ float s_mean[128];
  __shared__ float s_rstd[128];
  __bf16* s_y1 = (__bf16*)smem;
  float*  s_y2 = smem;

  int blk = blockIdx.x;
  int row0 = blk * 128;
  int tid = threadIdx.x;
  int w = tid >> 5;
  int lane = tid & 31;
  int mrow = row0 + w * 16;

  // --- Stage 1: y1 = relu(x @ W1), K=512, N=512 in two 256-col passes ---
  for (int pass = 0; pass < 2; ++pass) {
    v8f c[16] = {};
    v16bf a_cur = load_a_mlp(h3d, h_prime, mrow, 0);
    v16bf a_nxt = a_cur;
    v16bf b_cur[4], b_nxt[4];
    #pragma unroll
    for (int t = 0; t < 4; ++t)
      b_cur[t] = load_b_pack(W1pack, 0, pass * 16 + t, 32);

    #pragma unroll
    for (int gi = 0; gi < 64; ++gi) {        // 16 kt x 4 groups
      const int kt = gi >> 2, g = gi & 3;
      if (g == 0 && kt + 1 < 16)
        a_nxt = load_a_mlp(h3d, h_prime, mrow, kt + 1);
      if (gi + 1 < 64) {
        const int nkt = (g == 3) ? kt + 1 : kt;
        const int ng  = (g == 3) ? 0 : g + 1;
        #pragma unroll
        for (int t = 0; t < 4; ++t)
          b_nxt[t] = load_b_pack(W1pack, nkt, pass * 16 + ng * 4 + t, 32);
      }
      #pragma unroll
      for (int t = 0; t < 4; ++t)
        c[g * 4 + t] = __builtin_amdgcn_wmma_f32_16x16x32_bf16(
            false, a_cur, false, b_cur[t], (short)0, c[g * 4 + t], false, false);
      #pragma unroll
      for (int t = 0; t < 4; ++t) b_cur[t] = b_nxt[t];
      if (g == 3) a_cur = a_nxt;
    }

    int rl = w * 16 + 8 * (lane >> 4);
    int cl = pass * 256 + (lane & 15);
    #pragma unroll
    for (int t = 0; t < 16; ++t)
      #pragma unroll
      for (int i = 0; i < 8; ++i) {
        float v = c[t][i];
        s_y1[(rl + i) * Y1S + cl + t * 16] = f2bf(v > 0.f ? v : 0.f);
      }
  }
  __syncthreads();

  // --- Stage 2: y2 = y1 @ W2, K=512, A frags from LDS (pipelined) -------
  v8f c2[16] = {};
  int arow = w * 16 + (lane & 15);
  int kb8  = (lane >> 4) << 3;
  v16bf a_cur = load_a_lds(s_y1, Y1S, arow, kb8, 0);
  v16bf a_nxt = a_cur;
  v16bf b_cur[4], b_nxt[4];
  #pragma unroll
  for (int t = 0; t < 4; ++t) b_cur[t] = load_b_pack(W2pack, 0, t, 16);

  #pragma unroll
  for (int gi = 0; gi < 64; ++gi) {          // 16 kt x 4 groups
    const int kt = gi >> 2, g = gi & 3;
    if (g == 0 && kt + 1 < 16)
      a_nxt = load_a_lds(s_y1, Y1S, arow, kb8, kt + 1);
    if (gi + 1 < 64) {
      const int nkt = (g == 3) ? kt + 1 : kt;
      const int ng  = (g == 3) ? 0 : g + 1;
      #pragma unroll
      for (int t = 0; t < 4; ++t)
        b_nxt[t] = load_b_pack(W2pack, nkt, ng * 4 + t, 16);
    }
    #pragma unroll
    for (int t = 0; t < 4; ++t)
      c2[g * 4 + t] = __builtin_amdgcn_wmma_f32_16x16x32_bf16(
          false, a_cur, false, b_cur[t], (short)0, c2[g * 4 + t], false, false);
    #pragma unroll
    for (int t = 0; t < 4; ++t) b_cur[t] = b_nxt[t];
    if (g == 3) a_cur = a_nxt;
  }

  // Each wave only ever read its own 16 y1 rows; overwrite them with y2 f32.
  int rl2 = w * 16 + 8 * (lane >> 4);
  int cl2 = lane & 15;
  #pragma unroll
  for (int t = 0; t < 16; ++t)
    #pragma unroll
    for (int i = 0; i < 8; ++i)
      s_y2[(rl2 + i) * Y2S + t * 16 + cl2] = c2[t][i];
  __syncthreads();

  // --- Stage 3a: per-row mean / rstd (biased var, eps=1e-5) -------------
  if (tid < 128) {
    float s = 0.f, sq = 0.f;
    for (int cc = 0; cc < 256; ++cc) {
      float v = s_y2[tid * Y2S + cc];
      s += v; sq += v * v;
    }
    float mean = s * (1.f / 256.f);
    float var  = sq * (1.f / 256.f) - mean * mean;
    s_mean[tid] = mean;
    s_rstd[tid] = rsqrtf(var + IN_EPS);
  }
  __syncthreads();

  // --- Stage 3b: residual + write ---------------------------------------
  int r = tid >> 1;
  int cb = (tid & 1) * 128;
  float mean = s_mean[r], rstd = s_rstd[r];
  const float* hrow = h3d + (size_t)(row0 + r) * 256;
  float* orow = out + (size_t)(row0 + r) * 256;
  for (int cc = cb; cc < cb + 128; ++cc)
    orow[cc] = hrow[cc] + (s_y2[r * Y2S + cc] - mean) * rstd;
}

// ---------------------------------------------------------------------------
extern "C" void kernel_launch(void* const* d_in, const int* in_sizes, int n_in,
                              void* d_out, int out_size, void* d_ws, size_t ws_size,
                              hipStream_t stream) {
  const float* h2d = (const float*)d_in[0];
  const float* h3d = (const float*)d_in[1];
  const float* W   = (const float*)d_in[2];
  const float* a   = (const float*)d_in[3];
  const float* W1  = (const float*)d_in[4];
  const float* W2  = (const float*)d_in[5];
  float* out = (float*)d_out;
  (void)n_in; (void)out_size; (void)ws_size;

  int nodes = in_sizes[1] / 256;             // b * n1 = 16384

  char* ws = (char*)d_ws;
  __bf16* Wpack   = (__bf16*)(ws);                                   // 128 KB
  __bf16* W1pack  = (__bf16*)(ws + 131072);                          // 512 KB
  __bf16* W2pack  = (__bf16*)(ws + 131072 + 524288);                 // 256 KB
  float*  Wa2     = (float*) (ws + 131072 + 524288 + 262144);        // 1 KB
  float*  h_prime = (float*) (ws + 131072 + 524288 + 262144 + 4096); // 16 MB

  repack_b_kernel<<<(256 * 256 + 255) / 256, 256, 0, stream>>>(W,  Wpack,  256, 256);
  repack_b_kernel<<<(512 * 512 + 255) / 256, 256, 0, stream>>>(W1, W1pack, 512, 512);
  repack_b_kernel<<<(512 * 256 + 255) / 256, 256, 0, stream>>>(W2, W2pack, 512, 256);
  wa2_kernel<<<1, 256, 0, stream>>>(W, a, Wa2);

  gat_attn_kernel<<<nodes / 16, 256, 0, stream>>>(h2d, h3d, Wpack, a, Wa2, h_prime);
  gat_mlp_kernel<<<nodes / 128, 256, 0, stream>>>(h3d, h_prime, W1pack, W2pack, out);
}
